// DRQConv2d_10574209483146
// MI455X (gfx1250) — compile-verified, matrix operand order checked by reading, not run on hardware
//
#include <hip/hip_runtime.h>
#include <stdint.h>

typedef int v8i __attribute__((ext_vector_type(8)));

#define B_   32
#define C_   128
#define H_   56
#define W_   56
#define HP_  58
#define WP_  58
#define OC_  128
#define KTOT 1152     // C*3*3
#define KB_  18       // KTOT / 64
#define NT_  8        // OC / 16
#define EPS_ 1e-5f

// ---------------- zero fill (padded act buffers) ----------------
__global__ void fill_zero_u32(uint32_t* __restrict__ p, int n) {
    int i = blockIdx.x * blockDim.x + threadIdx.x;
    if (i < n) p[i] = 0u;
}

// ---------------- 8x8 avg-pool -> threshold mask ----------------
__global__ void pool_mask_kernel(const float* __restrict__ x, uint8_t* __restrict__ mask) {
    int i = blockIdx.x * blockDim.x + threadIdx.x;   // 32*128*7*7 = 200704
    if (i >= B_ * C_ * 7 * 7) return;
    int wq = i % 7, hq = (i / 7) % 7, c = (i / 49) % C_, b = i / (49 * C_);
    const float* base = x + (((size_t)(b * C_ + c)) * H_ + hq * 8) * W_ + wq * 8;
    float s = 0.f;
    for (int r = 0; r < 8; ++r) {
        const float* row = base + r * W_;
        #pragma unroll
        for (int q = 0; q < 8; ++q) s += row[q];
    }
    mask[i] = ((s * (1.0f / 64.0f)) >= 0.05f) ? 1 : 0;
}

// ---------------- activation fake-quant -> padded NHWC uint8 ----------------
__global__ void act_quant_kernel(const float* __restrict__ x,
                                 const uint8_t* __restrict__ mask,
                                 const float* __restrict__ sHp,
                                 const float* __restrict__ sLp,
                                 uint8_t* __restrict__ qh,
                                 uint8_t* __restrict__ ql) {
    int i = blockIdx.x * blockDim.x + threadIdx.x;   // NCHW linear, 12845056
    if (i >= B_ * C_ * H_ * W_) return;
    int w = i % W_, h = (i / W_) % H_, c = (i / (H_ * W_)) % C_, b = i / (C_ * H_ * W_);
    float xv = x[i];
    int m = mask[((b * C_ + c) * 7 + (h >> 3)) * 7 + (w >> 3)];
    float sH = sHp[0], sL = sLp[0];
    float vh = m ? xv : EPS_;        // masked_fill(~mask, eps)
    float vl = m ? EPS_ : xv;        // masked_fill(mask, eps)
    int qhv = (int)rintf(vh / sH);   // RNE like jnp.round
    qhv = qhv < 0 ? 0 : (qhv > 255 ? 255 : qhv);
    int qlv = (int)rintf(vl / sL);
    qlv = qlv < 0 ? 0 : (qlv > 15 ? 15 : qlv);
    size_t dst = (((size_t)b * HP_ + (h + 1)) * WP_ + (w + 1)) * C_ + c;
    qh[dst] = (uint8_t)qhv;
    ql[dst] = (uint8_t)qlv;
}

// ---------------- weight fake-quant + pack into WMMA B-operand layout ----------------
// B-matrix 64x16 IU8 layout (wave32): lane = half*16 + n (n = output ch within tile),
// VGPR v holds K = (v/2)*16 + half*8 + (v%2)*4 .. +3 for column n.
__global__ void wquant_pack_kernel(const float* __restrict__ wHigh,
                                   const float* __restrict__ wLow,
                                   const float* __restrict__ aHp,
                                   const float* __restrict__ aLp,
                                   int8_t* __restrict__ bpH,
                                   int8_t* __restrict__ bpL,
                                   float* __restrict__ sclH,
                                   float* __restrict__ sclL) {
    __shared__ float red[256];
    int oc = blockIdx.x;
    int tid = threadIdx.x;
    for (int path = 0; path < 2; ++path) {
        const float* wsrc = path ? wLow : wHigh;
        int8_t* bp = path ? bpL : bpH;
        float nmax = path ? 7.0f : 127.0f;
        float mv = 0.f;
        for (int k = tid; k < KTOT; k += 256)
            mv = fmaxf(mv, fabsf(wsrc[(size_t)oc * KTOT + k]));
        red[tid] = mv;
        __syncthreads();
        for (int off = 128; off > 0; off >>= 1) {
            if (tid < off) red[tid] = fmaxf(red[tid], red[tid + off]);
            __syncthreads();
        }
        float s = red[0] / nmax;
        if (!(s > 0.f)) s = 1.0f;
        float inv = 1.0f / s;
        int n = (int)nmax;
        for (int k = tid; k < KTOT; k += 256) {
            // K ordering: K = tap*128 + c  (tap = kh*3+kw); OIHW flat index
            int c = k & 127;
            int tap = k >> 7;
            int kh = tap / 3, kw = tap - kh * 3;
            float wv = wsrc[(((size_t)oc * C_ + c) * 3 + kh) * 3 + kw];
            int q = (int)rintf(wv * inv);
            q = q < -n ? -n : (q > n ? n : q);
            // pack position
            int nt = oc >> 4, nn = oc & 15;
            int kb = k >> 6, kin = k & 63;
            int byteo = kin & 3;
            int t = kin >> 2;                 // 0..15 within 64-K block
            int vv = ((t >> 2) << 1) | (t & 1);
            int halfp = (t >> 1) & 1;
            int lane = halfp * 16 + nn;
            size_t dword = ((size_t)(nt * KB_ + kb) * 32 + lane) * 8 + vv;
            bp[dword * 4 + byteo] = (int8_t)q;
        }
        if (tid == 0) {
            float a = path ? aLp[0] : aHp[0];
            (path ? sclL : sclH)[oc] = s * a;   // combined act*weight scale per OC
        }
        __syncthreads();
    }
}

// ---------------- main implicit-GEMM conv via V_WMMA_I32_16X16X64_IU8 ----------------
// 8 waves / block share the per-kb B slice via double-buffered LDS (2 x 16 KB).
__global__ __launch_bounds__(256)
void drq_conv_wmma(const uint8_t* __restrict__ qh,
                   const uint8_t* __restrict__ ql,
                   const int8_t* __restrict__ bpH,
                   const int8_t* __restrict__ bpL,
                   const float* __restrict__ sclH,
                   const float* __restrict__ sclL,
                   float* __restrict__ out) {
    // [buf][path][nt][lane][j] : per kb slice = 2 paths * 8 nt * 32 lanes * 32B = 16 KB
    __shared__ uint4 ldsB[2][2][NT_][32][2];

    int tid  = threadIdx.x;
    int wave = tid >> 5;
    int lane = tid & 31;
    int half = lane >> 4;
    int r    = lane & 15;            // A row within M-tile / output channel within N-tile
    int mtile = blockIdx.x * 8 + wave;
    int m0 = mtile * 16;
    int m  = m0 + r;
    int b  = m / 3136;               // 3136 % 16 == 0: tile never crosses batch
    int sp = m - b * 3136;
    int h  = sp / 56;
    int w  = sp - h * 56;

    // staging decomposition for this thread: 1024 uint4 per kb slice, 4 per thread
    int st_path[4], st_nt[4], st_q[4];
    #pragma unroll
    for (int i = 0; i < 4; ++i) {
        int u = i * 256 + tid;       // 0..1023
        st_path[i] = u >> 9;
        int rem = u & 511;
        st_nt[i] = rem >> 6;
        st_q[i]  = rem & 63;         // uint4 index within 1KB (nt,kb) chunk
    }
    const uint4* gB0 = (const uint4*)bpH;
    const uint4* gB1 = (const uint4*)bpL;

    v8i zero = {0, 0, 0, 0, 0, 0, 0, 0};
    v8i accH[NT_], accL[NT_];
    #pragma unroll
    for (int nt = 0; nt < NT_; ++nt) { accH[nt] = zero; accL[nt] = zero; }

    // prefetch kb = 0
    uint4 regs[4];
    #pragma unroll
    for (int i = 0; i < 4; ++i) {
        const uint4* src = st_path[i] ? gB1 : gB0;
        regs[i] = src[(size_t)(st_nt[i] * KB_ + 0) * 64 + st_q[i]];
    }

    for (int kb = 0; kb < KB_; ++kb) {
        int buf = kb & 1;
        // commit staged regs to LDS
        #pragma unroll
        for (int i = 0; i < 4; ++i)
            (&ldsB[buf][st_path[i]][st_nt[i]][0][0])[st_q[i]] = regs[i];

        // prefetch next kb slice while this one is consumed
        if (kb + 1 < KB_) {
            #pragma unroll
            for (int i = 0; i < 4; ++i) {
                const uint4* src = st_path[i] ? gB1 : gB0;
                regs[i] = src[(size_t)(st_nt[i] * KB_ + (kb + 1)) * 64 + st_q[i]];
            }
        }
        __syncthreads();   // B slice visible to all waves

        // ---- A operand loads (global, padded NHWC uint8) ----
        int tap = kb >> 1;
        int dh = tap / 3, dw = tap - dh * 3;
        // 8-bit A-matrix layout: lane holds row M=r, K = (v>>1)*16 + half*8 + (v&1)*4
        size_t aoff = (((size_t)b * HP_ + (h + dh)) * WP_ + (w + dw)) * C_
                      + (size_t)(kb & 1) * 64 + (size_t)half * 8;
        const uint2* pah = (const uint2*)(qh + aoff);
        const uint2* pal = (const uint2*)(ql + aoff);
        v8i Ah, Al;
        #pragma unroll
        for (int j = 0; j < 4; ++j) {          // K chunks at +0,+16,+32,+48 bytes
            uint2 th = pah[j * 2];
            uint2 tl = pal[j * 2];
            Ah[2 * j] = (int)th.x; Ah[2 * j + 1] = (int)th.y;
            Al[2 * j] = (int)tl.x; Al[2 * j + 1] = (int)tl.y;
        }

        // ---- 16 WMMAs against LDS-resident B ----
        #pragma unroll
        for (int nt = 0; nt < NT_; ++nt) {
            uint4 b0 = ldsB[buf][0][nt][lane][0];
            uint4 b1 = ldsB[buf][0][nt][lane][1];
            uint4 c0 = ldsB[buf][1][nt][lane][0];
            uint4 c1 = ldsB[buf][1][nt][lane][1];
            v8i Bh = {(int)b0.x, (int)b0.y, (int)b0.z, (int)b0.w,
                      (int)b1.x, (int)b1.y, (int)b1.z, (int)b1.w};
            v8i Bl = {(int)c0.x, (int)c0.y, (int)c0.z, (int)c0.w,
                      (int)c1.x, (int)c1.y, (int)c1.z, (int)c1.w};
            // A unsigned (uint8/uint4 codes), B signed (int8/int4 codes)
            accH[nt] = __builtin_amdgcn_wmma_i32_16x16x64_iu8(
                false, Ah, true, Bh, accH[nt], false, false);
            accL[nt] = __builtin_amdgcn_wmma_i32_16x16x64_iu8(
                false, Al, true, Bl, accL[nt], false, false);
        }
        __syncthreads();   // all reads of this buffer done before it is overwritten
    }

    // C/D layout: lane -> N = r, M = half*8 + g  => 8 consecutive spatial positions
    int sbase = (m0 - b * 3136) + half * 8;
    #pragma unroll
    for (int nt = 0; nt < NT_; ++nt) {
        int oc = nt * 16 + r;
        float sh = sclH[oc];
        float sl = sclL[oc];
        float4 f0, f1;
        f0.x = (float)accH[nt][0] * sh + (float)accL[nt][0] * sl;
        f0.y = (float)accH[nt][1] * sh + (float)accL[nt][1] * sl;
        f0.z = (float)accH[nt][2] * sh + (float)accL[nt][2] * sl;
        f0.w = (float)accH[nt][3] * sh + (float)accL[nt][3] * sl;
        f1.x = (float)accH[nt][4] * sh + (float)accL[nt][4] * sl;
        f1.y = (float)accH[nt][5] * sh + (float)accL[nt][5] * sl;
        f1.z = (float)accH[nt][6] * sh + (float)accL[nt][6] * sl;
        f1.w = (float)accH[nt][7] * sh + (float)accL[nt][7] * sl;
        size_t obase = ((size_t)(b * OC_ + oc)) * 3136 + sbase;
        *(float4*)(out + obase)     = f0;
        *(float4*)(out + obase + 4) = f1;
    }
}

extern "C" void kernel_launch(void* const* d_in, const int* in_sizes, int n_in,
                              void* d_out, int out_size, void* d_ws, size_t ws_size,
                              hipStream_t stream) {
    const float* x  = (const float*)d_in[0];
    const float* wH = (const float*)d_in[1];
    const float* wL = (const float*)d_in[2];
    const float* aH = (const float*)d_in[3];   // act_scale_high (1 elem)
    const float* aL = (const float*)d_in[4];   // act_scale_low  (1 elem)
    float* out = (float*)d_out;

    uint8_t* ws = (uint8_t*)d_ws;
    const size_t sz_q = (size_t)B_ * HP_ * WP_ * C_;        // 13,778,944 B
    uint8_t* qh  = ws;
    uint8_t* ql  = ws + sz_q;
    int8_t*  bpH = (int8_t*)(ws + 2 * sz_q);
    int8_t*  bpL = bpH + (size_t)OC_ * KTOT;                // 147,456 B each
    float*   sclH = (float*)(bpL + (size_t)OC_ * KTOT);
    float*   sclL = sclH + OC_;
    uint8_t* mask = (uint8_t*)(sclL + OC_);                 // 200,704 B

    int nfill = (int)(2 * sz_q / 4);
    fill_zero_u32<<<(nfill + 255) / 256, 256, 0, stream>>>((uint32_t*)ws, nfill);

    int npool = B_ * C_ * 7 * 7;
    pool_mask_kernel<<<(npool + 255) / 256, 256, 0, stream>>>(x, mask);

    int nact = B_ * C_ * H_ * W_;
    act_quant_kernel<<<(nact + 255) / 256, 256, 0, stream>>>(x, mask, aH, aL, qh, ql);

    wquant_pack_kernel<<<OC_, 256, 0, stream>>>(wH, wL, aH, aL, bpH, bpL, sclH, sclL);

    // M = 100352 positions -> 6272 tiles of 16 -> 784 blocks * 8 waves
    drq_conv_wmma<<<784, 256, 0, stream>>>(qh, ql, bpH, bpL, sclH, sclL, out);
}
